// Transformer_55946243998245
// MI455X (gfx1250) — matrix operand order repaired and checked by reading, not compiled, not run on hardware
//
#include <hip/hip_runtime.h>

// ---------------------------------------------------------------------------
// CDNA5 (gfx1250) transformer forward.
// All matmuls: v_wmma_f32_16x16x32_bf16, wave32.
// bf16 operands are prepared OUTSIDE the GEMM (weights pre-converted and
// pre-transposed to [N][K]; activations converted once when produced), so the
// GEMM hot loop is: async 16B copies -> LDS -> fragment gather -> WMMA.
// ---------------------------------------------------------------------------

typedef __attribute__((ext_vector_type(16))) __bf16 v16bf;
typedef __attribute__((ext_vector_type(8)))  float  v8f;
typedef int v4i_vs __attribute__((vector_size(16)));   // matches builtin proto

#define TD   512          // model dim D
#define THH  8            // heads
#define TDK  64           // head dim
#define TFF  2048         // ffn dim
#define TL   6            // layers
#define TV   32000        // vocab
#define TB   2            // batch
#define TS   512          // seq len
#define TBS  (TB * TS)    // tokens

// async global->LDS path (gfx1250): guarded so absence of the builtin still
// compiles via the plain-copy fallback.
#if defined(__HIP_DEVICE_COMPILE__) && defined(__gfx1250__) && \
    __has_builtin(__builtin_amdgcn_global_load_async_to_lds_b128) && \
    __has_builtin(__builtin_amdgcn_s_wait_asynccnt)
#define HAVE_ASYNC 1
#else
#define HAVE_ASYNC 0
#endif

__device__ __forceinline__ unsigned short f32_to_bf16_bits(float f) {
  unsigned int u = __float_as_uint(f);
  unsigned int r = u + 0x7FFFu + ((u >> 16) & 1u);   // round-to-nearest-even
  return (unsigned short)(r >> 16);
}
__device__ __forceinline__ __bf16 bf16_from_bits(unsigned short s) {
  union { unsigned short u; __bf16 h; } c; c.u = s; return c.h;
}

#if HAVE_ASYNC
__device__ __forceinline__ void async_cp16B(const unsigned short* g, unsigned short* l) {
  __builtin_amdgcn_global_load_async_to_lds_b128(
      (v4i_vs*)const_cast<unsigned short*>(g), (v4i_vs*)l, 0, 0);
}
#endif

// ---------------------------------------------------------------------------
// Batched bf16 WMMA GEMM:  C[z](MxN,f32) = act(scale * A[z](MxK) * B[z]^T + bias)
// A: bf16 row-major [M][K] (lda), B: bf16 [N][K] (ldb)  -- both contiguous in K.
// z = (bz,hz) = (z/Hdim, z%Hdim), with per-bz/hz pointer strides.
// mask_mode 1: byte mask[m*maskN+n] -> -inf (causal tgt_mask)
// mask_mode 2: byte mask[bz*maskN+n] -> -inf (src key mask)
// Block tile BM x BN; WGM x WGN waves; per-wave (BM/WGM) x (BN/WGN).
// ---------------------------------------------------------------------------
template <int BM, int BN, int WGM, int WGN>
__global__ __launch_bounds__(WGM * WGN * 32) void k_gemm_bf16(
    const unsigned short* __restrict__ A, int lda, long long Ab, long long Ah,
    const unsigned short* __restrict__ B, int ldb, long long Bb, long long Bh,
    float* __restrict__ C, int ldc, long long Cb, long long Ch,
    int M, int N, int K, int Hdim, float scale,
    const float* __restrict__ bias, int act,
    const unsigned char* __restrict__ mask, int mask_mode, int maskN)
{
  constexpr int T  = WGM * WGN * 32;
  constexpr int FM = BM / WGM / 16;
  constexpr int FN = BN / WGN / 16;
  static_assert(BM == BN && T == 2 * BM, "staging map assumes T == 2*BM == 2*BN");
  (void)M; (void)N;

  const int z = blockIdx.z, bz = z / Hdim, hz = z % Hdim;
  A += (long long)bz * Ab + (long long)hz * Ah;
  B += (long long)bz * Bb + (long long)hz * Bh;
  C += (long long)bz * Cb + (long long)hz * Ch;

  const int nbase = blockIdx.x * BN;
  const int mbase = blockIdx.y * BM;

  __shared__ unsigned short As[BM][40];   // 80B row pitch: 16B-aligned, bank-clean
  __shared__ unsigned short Bs[BN][40];

  const int t = threadIdx.x, lane = t & 31, wv = t >> 5;
  const int wr = wv / WGN, wc = wv % WGN;
  const int m16 = lane & 15, half = lane >> 4;
  const int srow = t >> 1, scol = (t & 1) * 16;   // each thread: 16 contiguous k

  v8f acc[FM][FN];
#pragma unroll
  for (int i = 0; i < FM; ++i)
#pragma unroll
    for (int j = 0; j < FN; ++j)
#pragma unroll
      for (int r = 0; r < 8; ++r) acc[i][j][r] = 0.0f;

  for (int k0 = 0; k0 < K; k0 += 32) {
    const unsigned short* ag = &A[(long long)(mbase + srow) * lda + k0 + scol];
    const unsigned short* bg = &B[(long long)(nbase + srow) * ldb + k0 + scol];
    if (k0 + 32 < K) {                       // prefetch next K tile
      __builtin_prefetch(ag + 32, 0, 1);
      __builtin_prefetch(bg + 32, 0, 1);
    }
#if HAVE_ASYNC
    async_cp16B(ag,     &As[srow][scol]);
    async_cp16B(ag + 8, &As[srow][scol + 8]);
    async_cp16B(bg,     &Bs[srow][scol]);
    async_cp16B(bg + 8, &Bs[srow][scol + 8]);
    __builtin_amdgcn_s_wait_asynccnt(0);
    __syncthreads();
#else
#pragma unroll
    for (int j = 0; j < 16; ++j) {
      As[srow][scol + j] = ag[j];
      Bs[srow][scol + j] = bg[j];
    }
    __syncthreads();
#endif

    // fragments: contiguous k order (k = 16*half + e) for both A and B
    v16bf af[FM], bf[FN];
#pragma unroll
    for (int fm = 0; fm < FM; ++fm) {
      int mrow = wr * (BM / WGM) + fm * 16 + m16;
#pragma unroll
      for (int e = 0; e < 16; ++e) {
        int kk = half * 16 + e;
        af[fm][e] = bf16_from_bits(As[mrow][kk]);
      }
    }
#pragma unroll
    for (int fn = 0; fn < FN; ++fn) {
      int nrow = wc * (BN / WGN) + fn * 16 + m16;
#pragma unroll
      for (int e = 0; e < 16; ++e)
        bf[fn][e] = bf16_from_bits(Bs[nrow][half * 16 + e]);
    }

#pragma unroll
    for (int fm = 0; fm < FM; ++fm)
#pragma unroll
      for (int fn = 0; fn < FN; ++fn)
        acc[fm][fn] = __builtin_amdgcn_wmma_f32_16x16x32_bf16(
            false, af[fm], false, bf[fn], (short)0, acc[fm][fn], false, false);
    __syncthreads();
  }

  // epilogue: C/D layout m = r + 8*half, n = lane%16
#pragma unroll
  for (int fm = 0; fm < FM; ++fm) {
#pragma unroll
    for (int fn = 0; fn < FN; ++fn) {
#pragma unroll
      for (int r = 0; r < 8; ++r) {
        int m = mbase + wr * (BM / WGM) + fm * 16 + r + half * 8;
        int n = nbase + wc * (BN / WGN) + fn * 16 + m16;
        float v = acc[fm][fn][r] * scale;
        if (bias) v += bias[n];
        if (act) v = fmaxf(v, 0.0f);
        if (mask_mode == 1) {
          if (mask[(long long)m * maskN + n]) v = -INFINITY;
        } else if (mask_mode == 2) {
          if (mask[(long long)bz * maskN + n]) v = -INFINITY;
        }
        C[(long long)m * ldc + n] = v;
      }
    }
  }
}

// elementwise f32 -> bf16 ----------------------------------------------------
__global__ __launch_bounds__(256) void k_cvt(
    const float* __restrict__ in, unsigned short* __restrict__ out, long long n)
{
  long long i = (long long)blockIdx.x * 2048 + threadIdx.x;
#pragma unroll
  for (int j = 0; j < 8; ++j) {
    long long idx = i + (long long)j * 256;
    if (idx < n) out[idx] = f32_to_bf16_bits(in[idx]);
  }
}

// f32 [R][C] -> bf16 [C][R] (tiled transpose; R,C multiples of 32) -----------
__global__ __launch_bounds__(256) void k_cvt_t(
    const float* __restrict__ in, unsigned short* __restrict__ out, int R, int C)
{
  __shared__ float tile[32][33];
  int tx = threadIdx.x & 31, ty = threadIdx.x >> 5;
  int r0 = blockIdx.y * 32, c0 = blockIdx.x * 32;
#pragma unroll
  for (int j = 0; j < 32; j += 8)
    tile[ty + j][tx] = in[(long long)(r0 + ty + j) * C + (c0 + tx)];
  __syncthreads();
#pragma unroll
  for (int j = 0; j < 32; j += 8)
    out[(long long)(c0 + ty + j) * R + (r0 + tx)] =
        f32_to_bf16_bits(tile[tx][ty + j]);
}

// row softmax over 512 keys with clip(1e-9, 1.0); in place -------------------
__global__ __launch_bounds__(256) void k_softmax_clip(float* __restrict__ sc)
{
  float* p = sc + (long long)blockIdx.x * TS;
  __shared__ float red[256];
  int t = threadIdx.x;
  float a = p[t], b = p[t + 256];
  red[t] = fmaxf(a, b);
  __syncthreads();
  for (int s = 128; s > 0; s >>= 1) {
    if (t < s) red[t] = fmaxf(red[t], red[t + s]);
    __syncthreads();
  }
  float mx = red[0];
  __syncthreads();
  float e0 = __expf(a - mx), e1 = __expf(b - mx);
  red[t] = e0 + e1;
  __syncthreads();
  for (int s = 128; s > 0; s >>= 1) {
    if (t < s) red[t] += red[t + s];
    __syncthreads();
  }
  float inv = 1.0f / red[0];
  p[t]       = fminf(fmaxf(e0 * inv, 1e-9f), 1.0f);
  p[t + 256] = fminf(fmaxf(e1 * inv, 1e-9f), 1.0f);
}

// out = LayerNorm(r + s) * g + b ; s may be null (plain LN) ------------------
__global__ __launch_bounds__(256) void k_add_ln(
    const float* __restrict__ r, const float* __restrict__ s,
    const float* __restrict__ g, const float* __restrict__ bt,
    float* __restrict__ out)
{
  long long base = (long long)blockIdx.x * TD;
  int t = threadIdx.x;
  float a = r[base + t]       + (s ? s[base + t]       : 0.0f);
  float b = r[base + t + 256] + (s ? s[base + t + 256] : 0.0f);
  __shared__ float red[256];
  red[t] = a + b;
  __syncthreads();
  for (int k = 128; k > 0; k >>= 1) {
    if (t < k) red[t] += red[t + k];
    __syncthreads();
  }
  float mu = red[0] * (1.0f / (float)TD);
  __syncthreads();
  float da = a - mu, db = b - mu;
  red[t] = da * da + db * db;
  __syncthreads();
  for (int k = 128; k > 0; k >>= 1) {
    if (t < k) red[t] += red[t + k];
    __syncthreads();
  }
  float rs = rsqrtf(red[0] * (1.0f / (float)TD) + 1e-5f);
  out[base + t]       = da * rs * g[t]       + bt[t];
  out[base + t + 256] = db * rs * g[t + 256] + bt[t + 256];
}

// x[bs,:] = emb[tok[bs],:] * sqrt(D) + pe[bs%S,:] ----------------------------
__global__ __launch_bounds__(256) void k_embed(
    const int* __restrict__ tok, const float* __restrict__ emb,
    const float* __restrict__ pe, float* __restrict__ x)
{
  int bs = blockIdx.x;
  long long ob = (long long)bs * TD;
  long long eb = (long long)tok[bs] * TD;
  long long pb = (long long)(bs % TS) * TD;
  for (int i = threadIdx.x; i < TD; i += 256)
    x[ob + i] = emb[eb + i] * 22.62741699796952f + pe[pb + i];
}

// ---------------------------------------------------------------------------
extern "C" void kernel_launch(void* const* d_in, const int* in_sizes, int n_in,
                              void* d_out, int out_size, void* d_ws, size_t ws_size,
                              hipStream_t stream)
{
  (void)in_sizes; (void)n_in; (void)out_size;
  auto F = [&](int i) -> const float* { return (const float*)d_in[i]; };

  const int* src = (const int*)d_in[0];
  const int* tgt = (const int*)d_in[1];
  const unsigned char* smask = (const unsigned char*)d_in[2];  // (B,1,1,S) bool
  const unsigned char* tmask = (const unsigned char*)d_in[3];  // (S,S) bool
  const float* pe = F(4);

  // param flattening (insertion order): 5 src_emb, 6 tgt_emb, enc @7 (16/layer),
  // enc_ln @103, dec @105 (26/layer), dec_ln @261, out_w 263, out_b 264
  const int ENC0 = 7, ENCLN = 103, DEC0 = 105, DECLN = 261, OUTW = 263;

  // ---- workspace layout ----
  float* ws = (float*)d_ws;
  size_t off = 0;
  auto allocF = [&](size_t n) { float* p = ws + off; off += n; return p; };
  float* x   = allocF((size_t)TBS * TD);
  float* y   = allocF((size_t)TBS * TD);
  float* mem = allocF((size_t)TBS * TD);
  float* qb  = allocF((size_t)TBS * TD);
  float* kb  = allocF((size_t)TBS * TD);
  float* vb  = allocF((size_t)TBS * TD);
  float* ctx = allocF((size_t)TBS * TD);
  float* tmp = allocF((size_t)TBS * TD);
  float* ffh = allocF((size_t)TBS * TFF);
  float* sc  = allocF((size_t)TB * THH * TS * TS);

  unsigned short* us = (unsigned short*)(ws + off);
  size_t uoff = 0;
  auto allocU = [&](size_t n) { unsigned short* p = us + uoff; uoff += n; return p; };
  unsigned short* xb    = allocU((size_t)TBS * TD);   // current q-input (x or y)
  unsigned short* membf = allocU((size_t)TBS * TD);
  unsigned short* qbf   = allocU((size_t)TBS * TD);
  unsigned short* kbf   = allocU((size_t)TBS * TD);
  unsigned short* vt    = allocU((size_t)TBS * TD);   // [B][D][S] transposed V
  unsigned short* ctxbf = allocU((size_t)TBS * TD);
  unsigned short* ffhbf = allocU((size_t)TBS * TFF);
  unsigned short* Pbf   = allocU((size_t)TB * THH * TS * TS);
  const size_t WTOT = (size_t)TL * (4 * TD * TD + 2 * TD * TFF)      // enc
                    + (size_t)TL * (8 * TD * TD + 2 * TD * TFF)      // dec
                    + (size_t)TD * TV;                               // out
  unsigned short* wpool = us + uoff;
  if (off * sizeof(float) + (uoff + WTOT) * sizeof(unsigned short) > ws_size)
    return;  // insufficient scratch

  // ---- weight conversion: f32 [K][N] -> bf16 [N][K] ----
  size_t woff = 0;
  auto cvtW = [&](int idx, int Kd, int Nd) -> const unsigned short* {
    unsigned short* p = wpool + woff; woff += (size_t)Kd * Nd;
    k_cvt_t<<<dim3(Nd / 32, Kd / 32), 256, 0, stream>>>(F(idx), p, Kd, Nd);
    return p;
  };
  const unsigned short* encW[TL][6];
  for (int l = 0; l < TL; ++l) {
    int p = ENC0 + 16 * l;
    encW[l][0] = cvtW(p + 0, TD, TD);   // Wq
    encW[l][1] = cvtW(p + 2, TD, TD);   // Wk
    encW[l][2] = cvtW(p + 4, TD, TD);   // Wv
    encW[l][3] = cvtW(p + 6, TD, TD);   // Wo
    encW[l][4] = cvtW(p + 8, TD, TFF);  // W1
    encW[l][5] = cvtW(p + 10, TFF, TD); // W2
  }
  const unsigned short* decW[TL][10];
  for (int l = 0; l < TL; ++l) {
    int p = DEC0 + 26 * l;
    for (int j = 0; j < 8; ++j) decW[l][j] = cvtW(p + 2 * j, TD, TD); // self+cross qkvo
    decW[l][8] = cvtW(p + 16, TD, TFF);
    decW[l][9] = cvtW(p + 18, TFF, TD);
  }
  const unsigned short* outWt = cvtW(OUTW, TD, TV);

  // ---- launch helpers ----
  auto cvt = [&](const float* in, unsigned short* out, long long n) {
    k_cvt<<<(unsigned)((n + 2047) / 2048), 256, 0, stream>>>(in, out, n);
  };
  // plain linear on big 128x128 tiles
  auto linear = [&](const unsigned short* A, int lda, const unsigned short* W,
                    const float* bias, float* C, int M, int N, int K, int act) {
    k_gemm_bf16<128, 128, 4, 2><<<dim3(N / 128, M / 128, 1), 256, 0, stream>>>(
        A, lda, 0LL, 0LL, W, K, 0LL, 0LL, C, N, 0LL, 0LL,
        M, N, K, 1, 1.0f, bias, act, (const unsigned char*)nullptr, 0, 0);
  };
  auto add_ln = [&](const float* r, const float* s, const float* g,
                    const float* b, float* o) {
    k_add_ln<<<TBS, 256, 0, stream>>>(r, s, g, b, o);
  };

  // multi-head attention
  auto mha = [&](const unsigned short* qinb, const unsigned short* kvinb,
                 const unsigned short* Wq, const unsigned short* Wk,
                 const unsigned short* Wv, const unsigned short* Wo,
                 const float* bq, const float* bk, const float* bv, const float* bo,
                 const unsigned char* mask, int mmode, float* outb) {
    linear(qinb, TD, Wq, bq, qb, TBS, TD, TD, 0);
    linear(kvinb, TD, Wk, bk, kb, TBS, TD, TD, 0);
    linear(kvinb, TD, Wv, bv, vb, TBS, TD, TD, 0);
    cvt(qb, qbf, (long long)TBS * TD);
    cvt(kb, kbf, (long long)TBS * TD);
    for (int b = 0; b < TB; ++b)   // V: [S][D] -> [D][S] bf16 per batch
      k_cvt_t<<<dim3(TD / 32, TS / 32), 256, 0, stream>>>(
          vb + (size_t)b * TS * TD, vt + (size_t)b * TD * TS, TS, TD);
    // scores[z] (SxS) = Q_h K_h^T / 8  (+mask)
    k_gemm_bf16<128, 128, 4, 2><<<dim3(TS / 128, TS / 128, TB * THH), 256, 0, stream>>>(
        qbf, TD, (long long)TS * TD, (long long)TDK,
        kbf, TD, (long long)TS * TD, (long long)TDK,
        sc, TS, (long long)THH * TS * TS, (long long)TS * TS,
        TS, TS, TDK, THH, 0.125f, nullptr, 0, mask, mmode, TS);
    k_softmax_clip<<<TB * THH * TS, 256, 0, stream>>>(sc);
    cvt(sc, Pbf, (long long)TB * THH * TS * TS);
    // ctx_h (SxDK) = P V_h   (B operand = vt [D][S] slice -> [N=dk][K=s])
    k_gemm_bf16<64, 64, 2, 2><<<dim3(TDK / 64, TS / 64, TB * THH), 128, 0, stream>>>(
        Pbf, TS, (long long)THH * TS * TS, (long long)TS * TS,
        vt, TS, (long long)TD * TS, (long long)TDK * TS,
        ctx, TD, (long long)TS * TD, (long long)TDK,
        TS, TDK, TS, THH, 1.0f, nullptr, 0, (const unsigned char*)nullptr, 0, 0);
    cvt(ctx, ctxbf, (long long)TBS * TD);
    linear(ctxbf, TD, Wo, bo, outb, TBS, TD, TD, 0);
  };

  // ---------------- encoder ----------------
  k_embed<<<TBS, 256, 0, stream>>>(src, F(5), pe, x);
  for (int l = 0; l < TL; ++l) {
    int p = ENC0 + 16 * l;
    cvt(x, xb, (long long)TBS * TD);
    mha(xb, xb, encW[l][0], encW[l][1], encW[l][2], encW[l][3],
        F(p + 1), F(p + 3), F(p + 5), F(p + 7), smask, 2, tmp);
    add_ln(x, tmp, F(p + 12), F(p + 13), x);
    cvt(x, xb, (long long)TBS * TD);
    linear(xb, TD, encW[l][4], F(p + 9), ffh, TBS, TFF, TD, 1);   // relu
    cvt(ffh, ffhbf, (long long)TBS * TFF);
    linear(ffhbf, TFF, encW[l][5], F(p + 11), tmp, TBS, TD, TFF, 0);
    add_ln(x, tmp, F(p + 14), F(p + 15), x);
  }
  add_ln(x, nullptr, F(ENCLN), F(ENCLN + 1), mem);
  cvt(mem, membf, (long long)TBS * TD);

  // ---------------- decoder ----------------
  k_embed<<<TBS, 256, 0, stream>>>(tgt, F(6), pe, y);
  for (int l = 0; l < TL; ++l) {
    int p = DEC0 + 26 * l;
    cvt(y, xb, (long long)TBS * TD);
    mha(xb, xb, decW[l][0], decW[l][1], decW[l][2], decW[l][3],
        F(p + 1), F(p + 3), F(p + 5), F(p + 7), tmask, 1, tmp);   // causal self
    add_ln(y, tmp, F(p + 20), F(p + 21), y);
    cvt(y, xb, (long long)TBS * TD);
    mha(xb, membf, decW[l][4], decW[l][5], decW[l][6], decW[l][7],
        F(p + 9), F(p + 11), F(p + 13), F(p + 15), smask, 2, tmp); // cross
    add_ln(y, tmp, F(p + 22), F(p + 23), y);
    cvt(y, xb, (long long)TBS * TD);
    linear(xb, TD, decW[l][8], F(p + 17), ffh, TBS, TFF, TD, 1);   // relu
    cvt(ffh, ffhbf, (long long)TBS * TFF);
    linear(ffhbf, TFF, decW[l][9], F(p + 19), tmp, TBS, TD, TFF, 0);
    add_ln(y, tmp, F(p + 24), F(p + 25), y);
  }
  add_ln(y, nullptr, F(DECLN), F(DECLN + 1), y);

  // ---------------- output projection ----------------
  cvt(y, xb, (long long)TBS * TD);
  linear(xb, TD, outWt, F(OUTW + 1), (float*)d_out, TBS, TV, TD, 0);
}
